// SequenceMemoryEncoder_7748121002260
// MI455X (gfx1250) — compile-verified, matrix operand order checked by reading, not run on hardware
//
#include <hip/hip_runtime.h>

typedef __attribute__((ext_vector_type(16))) _Float16 v16h;
typedef __attribute__((ext_vector_type(8)))  _Float16 v8h;
typedef __attribute__((ext_vector_type(8)))  float    v8f;

#define FMIN_F (-3.402823466e+38f)

constexpr int B_ = 8, T_ = 8192, D_ = 512, NB_ = 256;
constexpr int TOPK_ = 64, RECENT_ = 128;
constexpr int L_ = 16;
constexpr int SRCN = 193;   // 128 recent + 64 selected + 1 global
constexpr int SRCP = 208;   // padded to 13 tiles of 16

// ---- workspace layout (bytes) ----
constexpr size_t OFF_WV   = 0;                                   // comp_wv  f16 swizzled
constexpr size_t OFF_WW   = OFF_WV   + (size_t)D_ * D_ * 2;      // comp_ww  f16 swizzled
constexpr size_t OFF_WK   = OFF_WW   + (size_t)D_ * D_ * 2;      // pool_k_w f16 swizzled
constexpr size_t OFF_WVP  = OFF_WK   + (size_t)D_ * D_ * 2;      // pool_v_w f16 swizzled
constexpr size_t OFF_BT   = OFF_WVP  + (size_t)D_ * D_ * 2;      // bt  [B][NB][D] f32
constexpr size_t OFF_GSUM = OFF_BT   + (size_t)B_ * NB_ * D_ * 4;// gsum [B][D] f32
constexpr size_t OFF_IDX  = OFF_GSUM + (size_t)B_ * D_ * 4;      // idxs [B][64] i32
constexpr size_t OFF_SELM = OFF_IDX  + (size_t)B_ * TOPK_ * 4;   // selmask [B][64] i32
constexpr size_t OFF_SMSK = OFF_SELM + (size_t)B_ * TOPK_ * 4;   // smask [B][SRCP] i32
constexpr size_t OFF_SRC  = OFF_SMSK + (size_t)B_ * SRCP * 4;    // src (mt) [B][SRCP][D] f32
constexpr size_t OFF_KB   = OFF_SRC  + (size_t)B_ * SRCP * D_ * 4;
constexpr size_t OFF_VV   = OFF_KB   + (size_t)B_ * SRCP * D_ * 4;
constexpr size_t OFF_PART = OFF_VV   + (size_t)B_ * SRCP * D_ * 4; // partial [B][32][D] f32

__device__ __forceinline__ int imin(int a, int b) { return a < b ? a : b; }
__device__ __forceinline__ int imax(int a, int b) { return a > b ? a : b; }

// ---------------------------------------------------------------------------
// Swizzle one fp32 [512,512] (k-major) weight matrix into f16 WMMA B-fragment
// order: frag index = ((nt*16 + ks)*32 + lane)*16 + e, so each lane's fragment
// is one contiguous 32-byte load in the GEMM kernels.
// lane holds column n = nt*16 + (lane&15); element e -> K = ks*32 + (e&7) +
// 16*(e>>3) + 8*(lane>>4)   (mirrors ISA 16-bit A-matrix layout).
// ---------------------------------------------------------------------------
__global__ void __launch_bounds__(256)
wswizzle_kernel(const float* __restrict__ w, _Float16* __restrict__ o) {
  int gid  = blockIdx.x * 256 + threadIdx.x;    // 0 .. 512*512-1
  int e    = gid & 15;
  int lane = (gid >> 4) & 31;
  int ks   = (gid >> 9) & 15;
  int nt   = gid >> 13;
  int n = nt * 16 + (lane & 15);
  int k = ks * 32 + (e & 7) + ((e >> 3) << 4) + (((lane >> 4) & 1) << 3);
  o[gid] = (_Float16)w[k * D_ + n];
}

// ---------------------------------------------------------------------------
// Deterministic masked global-mean (partial sums per 256-token chunk)
// ---------------------------------------------------------------------------
__global__ void __launch_bounds__(512)
gsum_partial_kernel(const float* __restrict__ tokens, const int* __restrict__ lengths,
                    float* __restrict__ partial) {
  int b  = blockIdx.x >> 5;
  int ch = blockIdx.x & 31;
  int d  = threadIdx.x;
  int len = lengths[b];
  int t0 = ch * 256;
  const float* tp = tokens + ((size_t)b * T_ + t0) * D_ + d;
  float acc = 0.f;
  for (int t = 0; t < 256; ++t)
    if (t0 + t < len) acc += tp[(size_t)t * D_];
  partial[((size_t)b * 32 + ch) * D_ + d] = acc;
}

__global__ void __launch_bounds__(512)
gsum_reduce_kernel(const float* __restrict__ partial, float* __restrict__ gsum) {
  int b = blockIdx.x, d = threadIdx.x;
  float a = 0.f;
  for (int ch = 0; ch < 32; ++ch) a += partial[((size_t)b * 32 + ch) * D_ + d];
  gsum[b * D_ + d] = a;
}

// ---------------------------------------------------------------------------
// Fused block compression: values/logits GEMMs (WMMA f16->f32), masked softmax
// over the 32 tokens per channel, weighted sum, rmsnorm -> bt[B][NB][D].
// One workgroup per (batch, block). 512 threads = 16 waves.
// Dynamic LDS: A f16 32x520 (33280B) + outV f32 32x516 (66048B) +
//              outL f32 32x516 (66048B) + red 16 f32 (64B) = 165440 B.
// ---------------------------------------------------------------------------
__global__ void __launch_bounds__(512)
block_compress_kernel(const float* __restrict__ tokens, const int* __restrict__ lengths,
                      const float* __restrict__ comp_bv, const float* __restrict__ comp_bw,
                      const float* __restrict__ comp_pos, const float* __restrict__ comp_norm_w,
                      const _Float16* __restrict__ wv16, const _Float16* __restrict__ ww16,
                      float* __restrict__ bt) {
  extern __shared__ char smem[];
  _Float16* aT  = (_Float16*)smem;                      // 32 x 520 f16
  float*    outV = (float*)(smem + 33280);              // 32 x 516 f32
  float*    outL = (float*)(smem + 33280 + 66048);      // 32 x 516 f32
  float*    red  = (float*)(smem + 33280 + 2 * 66048);  // 16 f32

  const int tid  = threadIdx.x;
  const int b    = blockIdx.x >> 8;
  const int nb   = blockIdx.x & 255;
  const int len  = lengths[b];
  const int row0 = nb * 32;

  // Stage token tile fp32 -> f16 LDS (coalesced per row).
  {
    const float* tp = tokens + ((size_t)b * T_ + row0) * D_ + tid;
    for (int r = 0; r < 32; ++r)
      aT[r * 520 + tid] = (_Float16)tp[(size_t)r * D_];
  }
  __syncthreads();

  const int wave = tid >> 5, lane = tid & 31;
  const int gemm = wave >> 3;       // 0: values (wv), 1: logits (ww)
  const int wg   = wave & 7;        // 4 n-tiles each
  const _Float16* wmat = gemm ? ww16 : wv16;

  const v8f vzero = {0.f, 0.f, 0.f, 0.f, 0.f, 0.f, 0.f, 0.f};
  v8f acc[2][4];
#pragma unroll
  for (int m = 0; m < 2; ++m)
#pragma unroll
    for (int j = 0; j < 4; ++j) acc[m][j] = vzero;

  for (int ks = 0; ks < 16; ++ks) {
    v16h a0, a1;
    {
      int kb = ks * 32 + ((lane >> 4) << 3);
      const v8h* p0 = (const v8h*)&aT[(lane & 15) * 520 + kb];
      const v8h* p1 = (const v8h*)&aT[((lane & 15) + 16) * 520 + kb];
      v8h lo0 = p0[0], hi0 = p0[2];
      v8h lo1 = p1[0], hi1 = p1[2];
#pragma unroll
      for (int e = 0; e < 8; ++e) {
        a0[e] = lo0[e]; a0[8 + e] = hi0[e];
        a1[e] = lo1[e]; a1[8 + e] = hi1[e];
      }
    }
#pragma unroll
    for (int j = 0; j < 4; ++j) {
      int nt = wg * 4 + j;
      v16h bf = *(const v16h*)&wmat[(size_t)(((nt * 16 + ks) * 32 + lane)) << 4];
      acc[0][j] = __builtin_amdgcn_wmma_f32_16x16x32_f16(false, a0, false, bf, (short)0, acc[0][j], false, false);
      acc[1][j] = __builtin_amdgcn_wmma_f32_16x16x32_f16(false, a1, false, bf, (short)0, acc[1][j], false, false);
    }
  }

  // Spill tiles to LDS with bias / pos / mask applied.
  float* outp = gemm ? outL : outV;
#pragma unroll
  for (int mt = 0; mt < 2; ++mt)
#pragma unroll
    for (int j = 0; j < 4; ++j) {
      int col = (wg * 4 + j) * 16 + (lane & 15);
      int rb  = mt * 16 + ((lane >> 4) << 3);
#pragma unroll
      for (int r = 0; r < 8; ++r) {
        int row = rb + r;
        float v = acc[mt][j][r];
        if (gemm == 0) {
          v += comp_bv[col];
        } else {
          v += comp_bw[col] + comp_pos[row * D_ + col];
          if (row0 + row >= len) v = FMIN_F;
        }
        outp[row * 516 + col] = v;
      }
    }
  __syncthreads();

  // Per-channel masked softmax over the 32 tokens + weighted sum of values.
  float mx = FMIN_F;
  for (int r = 0; r < 32; ++r) mx = fmaxf(mx, outL[r * 516 + tid]);
  float sw = 0.f, btd = 0.f;
  for (int r = 0; r < 32; ++r) {
    float e = __expf(outL[r * 516 + tid] - mx);
    if (row0 + r >= len) e = 0.f;
    sw  += e;
    btd += e * outV[r * 516 + tid];
  }
  btd /= fmaxf(sw, 1.1920929e-07f);

  // rmsnorm over D (cross-wave reduction).
  float sq = btd * btd;
#pragma unroll
  for (int m = 16; m; m >>= 1) sq += __shfl_xor(sq, m, 32);
  if (lane == 0) red[wave] = sq;
  __syncthreads();
  if (tid < 16) {
    float v = red[tid];
#pragma unroll
    for (int m = 8; m; m >>= 1) v += __shfl_xor(v, m, 32);
    if (tid == 0) red[0] = v;
  }
  __syncthreads();
  float scale = rsqrtf(red[0] * (1.f / 512.f) + 1e-6f);
  float o = btd * scale * comp_norm_w[tid];
  if (row0 >= len) o = 0.f;  // fully-padded block
  bt[((size_t)b * NB_ + nb) * D_ + tid] = o;
}

// ---------------------------------------------------------------------------
// Index scoring + deterministic top-64 (tie -> lowest index). One WG/batch.
// ---------------------------------------------------------------------------
__global__ void __launch_bounds__(256)
score_topk_kernel(const float* __restrict__ query, const int* __restrict__ lengths,
                  const float* __restrict__ idx_qd_w, const float* __restrict__ idx_qd_b,
                  const float* __restrict__ idx_qu_w, const float* __restrict__ idx_qu_b,
                  const float* __restrict__ idx_k_w,  const float* __restrict__ idx_k_b,
                  const float* __restrict__ idx_hw_w, const float* __restrict__ idx_hw_b,
                  const float* __restrict__ idx_qn_w, const float* __restrict__ idx_kn_w,
                  const float* __restrict__ bt, int* __restrict__ idxs, int* __restrict__ selm) {
  __shared__ float s_ql[64];
  __shared__ float s_qh[512];
  __shared__ float s_hw[8];
  __shared__ float s_sc[256];
  __shared__ float s_tmp;
  const int b = blockIdx.x, tid = threadIdx.x;
  const int len = lengths[b];
  const float* q = query + b * D_;

  if (tid < 64) {
    float a = idx_qd_b[tid];
    for (int k = 0; k < D_; ++k) a += q[k] * idx_qd_w[k * 64 + tid];
    s_ql[tid] = a;
  }
  if (tid >= 64 && tid < 72) {
    int h = tid - 64;
    float a = idx_hw_b[h];
    for (int k = 0; k < D_; ++k) a += q[k] * idx_hw_w[k * 8 + h];
    s_hw[h] = a;
  }
  __syncthreads();
  if (tid == 0) {
    float ss = 0.f;
    for (int i = 0; i < 64; ++i) ss += s_ql[i] * s_ql[i];
    s_tmp = rsqrtf(ss * (1.f / 64.f) + 1e-6f);
    float mx = s_hw[0];
    for (int h = 1; h < 8; ++h) mx = fmaxf(mx, s_hw[h]);
    float sm = 0.f;
    for (int h = 0; h < 8; ++h) { float e = __expf(s_hw[h] - mx); s_hw[h] = e; sm += e; }
    for (int h = 0; h < 8; ++h) s_hw[h] /= sm;
  }
  __syncthreads();
  if (tid < 64) s_ql[tid] = s_ql[tid] * s_tmp * idx_qn_w[tid];
  __syncthreads();
  for (int j = tid; j < 512; j += 256) {
    float a = idx_qu_b[j];
#pragma unroll
    for (int i = 0; i < 64; ++i) a += s_ql[i] * idx_qu_w[i * 512 + j];
    s_qh[j] = a;
  }
  __syncthreads();
  // keys + scores: thread n = block n
  {
    const int n = tid;
    float key[64];
#pragma unroll
    for (int i = 0; i < 64; ++i) key[i] = idx_k_b[i];
    const float* btr = bt + ((size_t)b * NB_ + n) * D_;
    for (int k = 0; k < D_; ++k) {
      float x = btr[k];
#pragma unroll
      for (int i = 0; i < 64; ++i) key[i] += x * idx_k_w[k * 64 + i];
    }
    float ss = 0.f;
#pragma unroll
    for (int i = 0; i < 64; ++i) ss += key[i] * key[i];
    float sc2 = rsqrtf(ss * (1.f / 64.f) + 1e-6f);
#pragma unroll
    for (int i = 0; i < 64; ++i) key[i] *= sc2 * idx_kn_w[i];
    float score = 0.f;
#pragma unroll
    for (int h = 0; h < 8; ++h) {
      float d = 0.f;
#pragma unroll
      for (int i = 0; i < 64; ++i) d += s_qh[h * 64 + i] * key[i];
      score += fmaxf(d, 0.f) * s_hw[h];
    }
    if (n * 32 >= len) score = FMIN_F;
    s_sc[n] = score;
  }
  __syncthreads();
  if (tid == 0) {
    for (int k = 0; k < TOPK_; ++k) {
      int am = 0; float mv = s_sc[0];
      for (int n = 1; n < 256; ++n)
        if (s_sc[n] > mv) { mv = s_sc[n]; am = n; }
      idxs[b * TOPK_ + k] = am;
      selm[b * TOPK_ + k] = (am * 32 >= len) ? 1 : 0;
      s_sc[am] = -__builtin_inff();
    }
  }
}

// ---------------------------------------------------------------------------
// Assemble src (= mt, already masked) [B][SRCP][D] and smask.
// grid (SRCP, B), 512 threads = one row per block.
// ---------------------------------------------------------------------------
__global__ void __launch_bounds__(512)
assemble_src_kernel(const float* __restrict__ tokens, const int* __restrict__ lengths,
                    const float* __restrict__ bt, const float* __restrict__ gsum,
                    const int* __restrict__ idxs, const int* __restrict__ selm,
                    float* __restrict__ src, int* __restrict__ smask) {
  const int j = blockIdx.x, b = blockIdx.y, d = threadIdx.x;
  const int len = lengths[b];
  const int cl = imin(imax(len, 0), T_);
  float val = 0.f;
  int m = 1;
  if (j < RECENT_) {
    int start = imax(cl - RECENT_, 0);
    int pos = imin(start + j, T_ - 1);
    m = (j >= imin(cl, RECENT_)) ? 1 : 0;
    if (!m) {
      float x = tokens[((size_t)b * T_ + pos) * D_ + d];
      int i = d >> 1;
      float ang = (float)j * __expf(-(2.f * (float)i / (float)D_) * 9.210340371976184f);
      float pe = (d & 1) ? __cosf(ang) : __sinf(ang);
      val = x + pe;
    }
  } else if (j < RECENT_ + TOPK_) {
    int k = j - RECENT_;
    int id = idxs[b * TOPK_ + k];
    m = selm[b * TOPK_ + k];
    if (!m) val = bt[((size_t)b * NB_ + id) * D_ + d];
  } else if (j == RECENT_ + TOPK_) {
    m = (cl < 1) ? 1 : 0;
    if (!m) val = gsum[b * D_ + d] / (float)imax(cl, 1);
  }
  src[((size_t)b * SRCP + j) * D_ + d] = val;
  if (d == 0) smask[b * SRCP + j] = m;
}

// ---------------------------------------------------------------------------
// Pooling K/V projections with WMMA. grid = B*13 row-tiles, 256 thr = 8 waves.
// ---------------------------------------------------------------------------
__global__ void __launch_bounds__(256)
pool_kv_kernel(const float* __restrict__ src, const int* __restrict__ smask,
               const _Float16* __restrict__ wk16, const _Float16* __restrict__ wvp16,
               const float* __restrict__ bk, const float* __restrict__ bv,
               float* __restrict__ kbuf, float* __restrict__ vvbuf) {
  extern __shared__ char smem[];
  _Float16* aT = (_Float16*)smem;  // 16 x 520 f16
  const int tid = threadIdx.x;
  const int b = blockIdx.x / 13, rt = blockIdx.x % 13;
  const float* sp = src + ((size_t)b * SRCP + rt * 16) * D_;
  for (int qq = 0; qq < 32; ++qq) {
    int flat = qq * 256 + tid;
    int r = flat >> 9, c = flat & 511;
    aT[r * 520 + c] = (_Float16)sp[(size_t)r * D_ + c];
  }
  __syncthreads();

  const int wave = tid >> 5, lane = tid & 31;
  const int gemm = wave >> 2;   // 0: k, 1: vv
  const int wg   = wave & 3;    // 8 n-tiles each
  const _Float16* wmat = gemm ? wvp16 : wk16;

  const v8f vzero = {0.f, 0.f, 0.f, 0.f, 0.f, 0.f, 0.f, 0.f};
  v8f acc[8];
#pragma unroll
  for (int j = 0; j < 8; ++j) acc[j] = vzero;

  for (int ks = 0; ks < 16; ++ks) {
    v16h a;
    {
      int kb = ks * 32 + ((lane >> 4) << 3);
      const v8h* p = (const v8h*)&aT[(lane & 15) * 520 + kb];
      v8h lo = p[0], hi = p[2];
#pragma unroll
      for (int e = 0; e < 8; ++e) { a[e] = lo[e]; a[8 + e] = hi[e]; }
    }
#pragma unroll
    for (int j = 0; j < 8; ++j) {
      int nt = wg * 8 + j;
      v16h bf = *(const v16h*)&wmat[(size_t)(((nt * 16 + ks) * 32 + lane)) << 4];
      acc[j] = __builtin_amdgcn_wmma_f32_16x16x32_f16(false, a, false, bf, (short)0, acc[j], false, false);
    }
  }

  const float* bias = gemm ? bv : bk;
  float* outb = gemm ? vvbuf : kbuf;
#pragma unroll
  for (int j = 0; j < 8; ++j) {
    int col = (wg * 8 + j) * 16 + (lane & 15);
    int rb  = (lane >> 4) << 3;
#pragma unroll
    for (int r = 0; r < 8; ++r) {
      int row = rt * 16 + rb + r;
      float v = acc[j][r] + bias[col];
      if (gemm == 1 && smask[b * SRCP + row]) v = 0.f;
      outb[((size_t)b * SRCP + row) * D_ + col] = v;
    }
  }
}

// ---------------------------------------------------------------------------
// Latent cross-attention + rmsnorm -> d_out. One WG/batch, 512 thr = 16 waves.
// ---------------------------------------------------------------------------
__global__ void __launch_bounds__(512)
pool_attn_kernel(const float* __restrict__ query, const float* __restrict__ pool_latents,
                 const float* __restrict__ pool_q_w, const float* __restrict__ pool_q_b,
                 const float* __restrict__ kbuf, const float* __restrict__ vvbuf,
                 const int* __restrict__ smask, const float* __restrict__ pool_norm_w,
                 float* __restrict__ out) {
  __shared__ float s_lq[L_ * D_];
  __shared__ float s_pw[L_ * SRCP];
  __shared__ float s_scale[L_];
  __shared__ int   s_allm;
  const int b = blockIdx.x, tid = threadIdx.x;
  const float* q = query + b * D_;

  float qd = pool_q_b[tid];
  for (int k = 0; k < D_; ++k) qd += q[k] * pool_q_w[k * D_ + tid];
#pragma unroll
  for (int l = 0; l < L_; ++l) s_lq[l * D_ + tid] = pool_latents[l * D_ + tid] + qd;
  if (tid == 0) {
    int a = 1;
    for (int n = 0; n < SRCN; ++n) a &= smask[b * SRCP + n];
    s_allm = a;
  }
  __syncthreads();
  const int allm = s_allm;
  const float invs = 0.04419417382415922f;  // 512^-0.5

  for (int jj = tid; jj < L_ * SRCP; jj += 512) {
    int l = jj / SRCP, n = jj % SRCP;
    float sc = FMIN_F;
    if (n < SRCN) {
      int msk = smask[b * SRCP + n] && !allm;
      if (!msk) {
        float dacc = 0.f;
        const float* kr = kbuf + ((size_t)b * SRCP + n) * D_;
        for (int dd = 0; dd < D_; ++dd) dacc += s_lq[l * D_ + dd] * kr[dd];
        sc = dacc * invs;
      }
    }
    s_pw[jj] = sc;
  }
  __syncthreads();

  {  // softmax per latent row (wave l handles row l)
    const int wave = tid >> 5, lane = tid & 31;
    float mx = -__builtin_inff();
    for (int n = lane; n < SRCP; n += 32) mx = fmaxf(mx, s_pw[wave * SRCP + n]);
#pragma unroll
    for (int m = 16; m; m >>= 1) mx = fmaxf(mx, __shfl_xor(mx, m, 32));
    float sm = 0.f;
    for (int n = lane; n < SRCP; n += 32) {
      float e = __expf(s_pw[wave * SRCP + n] - mx);
      s_pw[wave * SRCP + n] = e;
      sm += e;
    }
#pragma unroll
    for (int m = 16; m; m >>= 1) sm += __shfl_xor(sm, m, 32);
    float inv = 1.f / sm;
    for (int n = lane; n < SRCP; n += 32) s_pw[wave * SRCP + n] *= inv;
  }
  __syncthreads();

  float lat[L_];
#pragma unroll
  for (int l = 0; l < L_; ++l) {
    float a = 0.f;
    for (int n = 0; n < SRCN; ++n)
      a += s_pw[l * SRCP + n] * vvbuf[((size_t)b * SRCP + n) * D_ + tid];
    lat[l] = a;
  }
  __syncthreads();
#pragma unroll
  for (int l = 0; l < L_; ++l) s_lq[l * D_ + tid] = lat[l];
  __syncthreads();
  {
    const int wave = tid >> 5, lane = tid & 31;
    float ss = 0.f;
    for (int dd = lane; dd < D_; dd += 32) { float v = s_lq[wave * D_ + dd]; ss += v * v; }
#pragma unroll
    for (int m = 16; m; m >>= 1) ss += __shfl_xor(ss, m, 32);
    if (lane == 0) s_scale[wave] = rsqrtf(ss * (1.f / 512.f) + 1e-6f);
  }
  __syncthreads();
#pragma unroll
  for (int l = 0; l < L_; ++l) {
    float o = s_lq[l * D_ + tid] * s_scale[l] * pool_norm_w[tid];
    if (allm) o = 0.f;
    out[((size_t)b * L_ + l) * D_ + tid] = o;
  }
}

// ---------------------------------------------------------------------------
extern "C" void kernel_launch(void* const* d_in, const int* in_sizes, int n_in,
                              void* d_out, int out_size, void* d_ws, size_t ws_size,
                              hipStream_t stream) {
  (void)in_sizes; (void)n_in; (void)out_size; (void)ws_size;
  const float* tokens      = (const float*)d_in[0];
  const int*   lengths     = (const int*)  d_in[1];
  const float* query       = (const float*)d_in[2];
  const float* comp_wv     = (const float*)d_in[3];
  const float* comp_bv     = (const float*)d_in[4];
  const float* comp_ww     = (const float*)d_in[5];
  const float* comp_bw     = (const float*)d_in[6];
  const float* comp_pos    = (const float*)d_in[7];
  const float* comp_norm_w = (const float*)d_in[8];
  const float* idx_qd_w    = (const float*)d_in[9];
  const float* idx_qd_b    = (const float*)d_in[10];
  const float* idx_qu_w    = (const float*)d_in[11];
  const float* idx_qu_b    = (const float*)d_in[12];
  const float* idx_k_w     = (const float*)d_in[13];
  const float* idx_k_b     = (const float*)d_in[14];
  const float* idx_hw_w    = (const float*)d_in[15];
  const float* idx_hw_b    = (const float*)d_in[16];
  const float* idx_qn_w    = (const float*)d_in[17];
  const float* idx_kn_w    = (const float*)d_in[18];
  const float* pool_latents= (const float*)d_in[19];
  const float* pool_q_w    = (const float*)d_in[20];
  const float* pool_q_b    = (const float*)d_in[21];
  const float* pool_k_w    = (const float*)d_in[22];
  const float* pool_k_b    = (const float*)d_in[23];
  const float* pool_v_w    = (const float*)d_in[24];
  const float* pool_v_b    = (const float*)d_in[25];
  const float* pool_norm_w = (const float*)d_in[26];

  char* ws = (char*)d_ws;
  _Float16* wv16  = (_Float16*)(ws + OFF_WV);
  _Float16* ww16  = (_Float16*)(ws + OFF_WW);
  _Float16* wk16  = (_Float16*)(ws + OFF_WK);
  _Float16* wvp16 = (_Float16*)(ws + OFF_WVP);
  float* bt      = (float*)(ws + OFF_BT);
  float* gsum    = (float*)(ws + OFF_GSUM);
  int*   idxs    = (int*)  (ws + OFF_IDX);
  int*   selm    = (int*)  (ws + OFF_SELM);
  int*   smask   = (int*)  (ws + OFF_SMSK);
  float* src     = (float*)(ws + OFF_SRC);
  float* kbuf    = (float*)(ws + OFF_KB);
  float* vvbuf   = (float*)(ws + OFF_VV);
  float* partial = (float*)(ws + OFF_PART);

  // Allow >64KB dynamic LDS for the fused compression kernel (320KB/WGP).
  // Idempotent driver call, not a stream op: safe under graph capture, and
  // called unconditionally every launch (no static state).
  hipFuncSetAttribute((const void*)block_compress_kernel,
                      hipFuncAttributeMaxDynamicSharedMemorySize, 165440);

  // 1) weight swizzle fp32 -> f16 WMMA fragment order
  wswizzle_kernel<<<1024, 256, 0, stream>>>(comp_wv, wv16);
  wswizzle_kernel<<<1024, 256, 0, stream>>>(comp_ww, ww16);
  wswizzle_kernel<<<1024, 256, 0, stream>>>(pool_k_w, wk16);
  wswizzle_kernel<<<1024, 256, 0, stream>>>(pool_v_w, wvp16);

  // 2) deterministic masked global token mean
  gsum_partial_kernel<<<B_ * 32, 512, 0, stream>>>(tokens, lengths, partial);
  gsum_reduce_kernel<<<B_, 512, 0, stream>>>(partial, gsum);

  // 3) fused block compression (dominant WMMA kernel)
  block_compress_kernel<<<B_ * NB_, 512, 165440, stream>>>(
      tokens, lengths, comp_bv, comp_bw, comp_pos, comp_norm_w, wv16, ww16, bt);

  // 4) scoring + top-k
  score_topk_kernel<<<B_, 256, 0, stream>>>(
      query, lengths, idx_qd_w, idx_qd_b, idx_qu_w, idx_qu_b, idx_k_w, idx_k_b,
      idx_hw_w, idx_hw_b, idx_qn_w, idx_kn_w, bt, idxs, selm);

  // 5) assemble masked src
  assemble_src_kernel<<<dim3(SRCP, B_), 512, 0, stream>>>(
      tokens, lengths, bt, gsum, idxs, selm, src, smask);

  // 6) pooling K/V projections (WMMA)
  pool_kv_kernel<<<B_ * 13, 256, 16 * 520 * 2, stream>>>(
      src, smask, wk16, wvp16, pool_k_b, pool_v_b, kbuf, vvbuf);

  // 7) latent cross-attention + rmsnorm -> output
  pool_attn_kernel<<<B_, 512, 0, stream>>>(
      query, pool_latents, pool_q_w, pool_q_b, kbuf, vvbuf, smask, pool_norm_w,
      (float*)d_out);
}
